// MultiTaskVQAModel_15839839387823
// MI455X (gfx1250) — matrix operand order repaired and checked by reading, not compiled, not run on hardware
//
#include <hip/hip_runtime.h>
#include <hip/hip_bf16.h>
#include <math.h>
#include <stdint.h>

typedef unsigned short u16t;
typedef __attribute__((ext_vector_type(16))) __bf16 v16bf;
typedef __attribute__((ext_vector_type(8)))  float  v8f;

union Frag { uint4 u[2]; v16bf v; };

#define BM 128
#define BN 64
#define BK 32
#define LDSP 40   // padded LDS row stride in bf16 elements (80B -> conflict-free)

__device__ __forceinline__ u16t f2bf(float f) {
  unsigned u = __float_as_uint(f);
  u += 0x7FFFu + ((u >> 16) & 1u);   // round-to-nearest-even
  return (u16t)(u >> 16);
}
__device__ __forceinline__ float bf2f(u16t h) { return __uint_as_float(((unsigned)h) << 16); }

// ---- gfx1250 async global->LDS copy (ASYNCcnt-tracked, no VGPR round trip) ----
// LDS address for the VDST operand = low 32 bits of the generic pointer
// (flat LDS aperture, ISA 10.2).
__device__ __forceinline__ unsigned lds_addr32(const void* p) {
  return (unsigned)(unsigned long long)p;
}
__device__ __forceinline__ void async_cp16(unsigned lds, const void* g) {
  asm volatile("global_load_async_to_lds_b128 %0, %1, off" :: "v"(lds), "v"(g) : "memory");
}
// IOFFSET is added to BOTH the LDS and global address (ISA 15.18.3 pseudocode),
// so one base pair covers the second 16B half of a 32B segment.
__device__ __forceinline__ void async_cp16_plus16(unsigned lds, const void* g) {
  asm volatile("global_load_async_to_lds_b128 %0, %1, off offset:16" :: "v"(lds), "v"(g) : "memory");
}
__device__ __forceinline__ void wait_async0() {
  asm volatile("s_wait_asynccnt 0x0" ::: "memory");
}

// ---------------- f32 -> bf16 bulk conversion (all sizes are multiples of 4) ----------------
__global__ void cvt_f32_to_bf16(const float* __restrict__ s, u16t* __restrict__ d, long n4) {
  long i = (long)blockIdx.x * blockDim.x + threadIdx.x;
  const long stride = (long)gridDim.x * blockDim.x;
  for (; i < n4; i += stride) {
    float4 f = ((const float4*)s)[i];
    uint2 o;
    o.x = (unsigned)f2bf(f.x) | ((unsigned)f2bf(f.y) << 16);
    o.y = (unsigned)f2bf(f.z) | ((unsigned)f2bf(f.w) << 16);
    ((uint2*)d)[i] = o;
  }
}

__global__ void zero_f32(float* __restrict__ p, long n) {
  long i = (long)blockIdx.x * blockDim.x + threadIdx.x;
  const long stride = (long)gridDim.x * blockDim.x;
  for (; i < n; i += stride) p[i] = 0.0f;
}

// ---------------- generic bf16 WMMA GEMM: out = act(A @ W^T + bias) ----------------
// A: [M,K] bf16 row-major, W: [N,K] bf16 row-major, bias: [N] f32, out: [M,N] bf16.
// M must be a multiple of 128. N, K arbitrary (guarded).
template<bool TANH>
__global__ void __launch_bounds__(256) gemm_bf16_wmma(
    const u16t* __restrict__ A, const u16t* __restrict__ W,
    const float* __restrict__ bias, u16t* __restrict__ out,
    int M, int N, int K)
{
  __shared__ u16t As[BM * LDSP];
  __shared__ u16t Ws[BN * LDSP];
  const int tid   = threadIdx.x;
  const int mbase = blockIdx.x * BM;
  const int nbase = blockIdx.y * BN;
  const int wave = tid >> 5, lane = tid & 31;
  const int lm = lane & 15, lh = lane >> 4;
  const int mrow = wave * 16;
  const int arow = tid >> 1, aseg = tid & 1;   // A stage: 128 rows x 2 x 16 halves
  const int wrow = tid >> 2, wseg = tid & 3;   // W stage:  64 rows x 4 x  8 halves

  v8f acc[4] = {};

  for (int k0 = 0; k0 < K; k0 += BK) {
    { // stage A tile (BM x BK)
      int gk = k0 + aseg * 16;
      const u16t* src = A + (size_t)(mbase + arow) * K + gk;
      u16t* dst = &As[arow * LDSP + aseg * 16];
      if (gk + 16 <= K) {
        ((uint4*)dst)[0] = ((const uint4*)src)[0];
        ((uint4*)dst)[1] = ((const uint4*)src)[1];
      } else {
        #pragma unroll
        for (int i = 0; i < 16; ++i) dst[i] = (gk + i < K) ? src[i] : (u16t)0;
      }
    }
    { // stage W tile (BN x BK)
      int gn = nbase + wrow, gk = k0 + wseg * 8;
      const u16t* src = W + (size_t)gn * K + gk;
      u16t* dst = &Ws[wrow * LDSP + wseg * 8];
      if (gn < N && gk + 8 <= K) {
        *(uint4*)dst = *(const uint4*)src;
      } else {
        #pragma unroll
        for (int i = 0; i < 8; ++i) dst[i] = (gn < N && gk + i < K) ? src[i] : (u16t)0;
      }
    }
    __syncthreads();

    // A fragment per ISA layout: lanes 0-15 hold K 0-7 & 16-23, lanes 16-31 hold K 8-15 & 24-31
    Frag a;
    a.u[0] = *(const uint4*)&As[(mrow + lm) * LDSP + lh * 8];
    a.u[1] = *(const uint4*)&As[(mrow + lm) * LDSP + lh * 8 + 16];
    #pragma unroll
    for (int ns = 0; ns < 4; ++ns) {
      // B fragment: lanes 0-15 hold K 0-7 & 16-23, lanes 16-31 hold K 8-15 & 24-31 for column n=lm
      Frag b;
      b.u[0] = *(const uint4*)&Ws[(ns * 16 + lm) * LDSP + lh * 8];
      b.u[1] = *(const uint4*)&Ws[(ns * 16 + lm) * LDSP + lh * 8 + 16];
      acc[ns] = __builtin_amdgcn_wmma_f32_16x16x32_bf16(
          false, a.v, false, b.v, (short)0, acc[ns], false, false);
    }
    __syncthreads();
  }

  #pragma unroll
  for (int ns = 0; ns < 4; ++ns) {
    int gn = nbase + ns * 16 + lm;
    if (gn >= N) continue;
    float bb = bias[gn];
    #pragma unroll
    for (int i = 0; i < 8; ++i) {
      int gm = mbase + mrow + lh * 8 + i;
      float v = acc[ns][i] + bb;
      if (TANH) v = tanhf(v);
      out[(size_t)gm * N + gn] = f2bf(v);
    }
  }
}

// ---------------- fused Mutan: z = sum_r (h0@Wm0[r]^T + bm0)*(h1@Wm1[r]^T + bm1) ----------------
// H0,H1: [B,1600] bf16; Wm0,Wm1: [24000,1600] bf16; bm: [24000] f32; Z: [B,1600] bf16.
// Double-buffered async global->LDS staging (global_load_async_to_lds_b128 + s_wait_asynccnt),
// flattened over (rank=15) x (K=1600/32=50) = 750 pipeline stages.
__global__ void __launch_bounds__(256) mutan_fused_wmma(
    const u16t* __restrict__ H0, const u16t* __restrict__ H1,
    const u16t* __restrict__ Wm0, const u16t* __restrict__ Wm1,
    const float* __restrict__ bm0, const float* __restrict__ bm1,
    u16t* __restrict__ Z)
{
  const int KD = 1600;
  const int KSTEPS = 50;              // 1600 / 32
  const int NSTEP = 15 * KSTEPS;      // rank * ksteps
  __shared__ u16t As0[2][BM * LDSP], As1[2][BM * LDSP];
  __shared__ u16t Bs0[2][BN * LDSP], Bs1[2][BN * LDSP];
  const int tid   = threadIdx.x;
  const int mbase = blockIdx.x * BM;
  const int nbase = blockIdx.y * BN;
  const int wave = tid >> 5, lane = tid & 31;
  const int lm = lane & 15, lh = lane >> 4;
  const int mrow = wave * 16;
  const int arow = tid >> 1, aseg = tid & 1;
  const int wrow = tid >> 2, wseg = tid & 3;

  // issue this thread's 6 async 16B copies for pipeline stage `step` into buffer `buf`
  auto stage = [&](int step, int buf) {
    int r  = step / KSTEPS;
    int k0 = (step - r * KSTEPS) * BK;
    {
      int gk = k0 + aseg * 16;
      size_t off = (size_t)(mbase + arow) * KD + gk;
      unsigned l0 = lds_addr32(&As0[buf][arow * LDSP + aseg * 16]);
      unsigned l1 = lds_addr32(&As1[buf][arow * LDSP + aseg * 16]);
      async_cp16(l0, H0 + off);  async_cp16_plus16(l0, H0 + off);
      async_cp16(l1, H1 + off);  async_cp16_plus16(l1, H1 + off);
    }
    {
      int gk = k0 + wseg * 8;
      size_t off = (size_t)(r * 1600 + nbase + wrow) * KD + gk;
      async_cp16(lds_addr32(&Bs0[buf][wrow * LDSP + wseg * 8]), Wm0 + off);
      async_cp16(lds_addr32(&Bs1[buf][wrow * LDSP + wseg * 8]), Wm1 + off);
    }
  };

  v8f zacc[4] = {};
  v8f c0[4] = {}, c1[4] = {};

  stage(0, 0);
  for (int step = 0; step < NSTEP; ++step) {
    const int buf = step & 1;
    wait_async0();        // my async writes into `buf` are complete
    __syncthreads();      // everyone's writes visible; prior reads of buf^1 retired
    if (step + 1 < NSTEP) stage(step + 1, buf ^ 1);   // prefetch overlaps WMMA below

    Frag a0, a1;
    a0.u[0] = *(const uint4*)&As0[buf][(mrow + lm) * LDSP + lh * 8];
    a0.u[1] = *(const uint4*)&As0[buf][(mrow + lm) * LDSP + lh * 8 + 16];
    a1.u[0] = *(const uint4*)&As1[buf][(mrow + lm) * LDSP + lh * 8];
    a1.u[1] = *(const uint4*)&As1[buf][(mrow + lm) * LDSP + lh * 8 + 16];
    #pragma unroll
    for (int ns = 0; ns < 4; ++ns) {
      Frag b0, b1;
      b0.u[0] = *(const uint4*)&Bs0[buf][(ns * 16 + lm) * LDSP + lh * 8];
      b0.u[1] = *(const uint4*)&Bs0[buf][(ns * 16 + lm) * LDSP + lh * 8 + 16];
      b1.u[0] = *(const uint4*)&Bs1[buf][(ns * 16 + lm) * LDSP + lh * 8];
      b1.u[1] = *(const uint4*)&Bs1[buf][(ns * 16 + lm) * LDSP + lh * 8 + 16];
      c0[ns] = __builtin_amdgcn_wmma_f32_16x16x32_bf16(false, a0.v, false, b0.v, (short)0, c0[ns], false, false);
      c1[ns] = __builtin_amdgcn_wmma_f32_16x16x32_bf16(false, a1.v, false, b1.v, (short)0, c1[ns], false, false);
    }

    if ((step % KSTEPS) == (KSTEPS - 1)) {
      // rank-r epilogue: (c0+bm0)*(c1+bm1) accumulated in registers; m never hits memory
      const int r = step / KSTEPS;
      const int wbase = r * 1600 + nbase;
      #pragma unroll
      for (int ns = 0; ns < 4; ++ns) {
        int gw = wbase + ns * 16 + lm;
        float bb0 = bm0[gw], bb1 = bm1[gw];
        #pragma unroll
        for (int i = 0; i < 8; ++i) {
          zacc[ns][i] += (c0[ns][i] + bb0) * (c1[ns][i] + bb1);
          c0[ns][i] = 0.0f;
          c1[ns][i] = 0.0f;
        }
      }
    }
  }

  #pragma unroll
  for (int ns = 0; ns < 4; ++ns) {
    int gn = nbase + ns * 16 + lm;
    #pragma unroll
    for (int i = 0; i < 8; ++i) {
      int gm = mbase + mrow + lh * 8 + i;
      Z[(size_t)gm * 1600 + gn] = f2bf(zacc[ns][i]);
    }
  }
}

// ---------------- tiny per-question-type head: out[b, lo+j] = hid[b]·W2[j] + b2[j] (masked) ----
__global__ void classifier_head(const u16t* __restrict__ hid, const float* __restrict__ W2,
                                const float* __restrict__ b2, const int* __restrict__ qtype,
                                int qt, int lo, int na, float* __restrict__ out)
{
  int b = blockIdx.x;
  if (qtype[b] != qt) return;
  const u16t* h = hid + (size_t)b * 256;
  for (int j = threadIdx.x; j < na; j += blockDim.x) {
    const float* w = W2 + (size_t)j * 256;
    float s = b2[j];
    for (int k = 0; k < 256; ++k) s += bf2f(h[k]) * w[k];
    out[(size_t)b * 95 + lo + j] = s;
  }
}

// =====================================================================================
extern "C" void kernel_launch(void* const* d_in, const int* in_sizes, int n_in,
                              void* d_out, int out_size, void* d_ws, size_t ws_size,
                              hipStream_t stream)
{
  (void)n_in; (void)out_size; (void)ws_size;
  const int B = in_sizes[0] / 768;   // 4096 (multiple of 128)

  const float* input_v = (const float*)d_in[0];
  const float* input_q = (const float*)d_in[1];
  const int*   qtype   = (const int*)d_in[2];
  const float* Wv  = (const float*)d_in[3];   const float* bv  = (const float*)d_in[4];
  const float* Wq  = (const float*)d_in[5];   const float* bq  = (const float*)d_in[6];
  const float* W0  = (const float*)d_in[7];   const float* b0  = (const float*)d_in[8];
  const float* W1  = (const float*)d_in[9];   const float* b1  = (const float*)d_in[10];
  const float* Wm0 = (const float*)d_in[11];  const float* bm0 = (const float*)d_in[12];
  const float* Wm1 = (const float*)d_in[13];  const float* bm1 = (const float*)d_in[14];
  const float* Wo  = (const float*)d_in[15];  const float* bo  = (const float*)d_in[16];
  float* out = (float*)d_out;

  // deterministic bump allocator over workspace (every chunk 256B aligned)
  char* ws = (char*)d_ws;
  size_t off = 0;
  auto alloc16 = [&](size_t elems) -> u16t* {
    u16t* p = (u16t*)(ws + off);
    off = (off + elems * 2 + 255) & ~(size_t)255;
    return p;
  };

  u16t* v_bf   = alloc16((size_t)B * 768);
  u16t* q_bf   = alloc16((size_t)B * 2400);
  u16t* Wv_bf  = alloc16((size_t)1200 * 768);
  u16t* Wq_bf  = alloc16((size_t)1200 * 2400);
  u16t* W0_bf  = alloc16((size_t)1600 * 1200);
  u16t* W1_bf  = alloc16((size_t)1600 * 1200);
  u16t* Wm0_bf = alloc16((size_t)24000 * 1600);
  u16t* Wm1_bf = alloc16((size_t)24000 * 1600);
  u16t* Wo_bf  = alloc16((size_t)1200 * 1600);
  u16t* cW1_bf[4];
  for (int q = 0; q < 4; ++q) cW1_bf[q] = alloc16((size_t)256 * 1200);
  u16t* xv_bf  = alloc16((size_t)B * 1200);
  u16t* xq_bf  = alloc16((size_t)B * 1200);
  u16t* h0_bf  = alloc16((size_t)B * 1600);
  u16t* h1_bf  = alloc16((size_t)B * 1600);
  u16t* z_bf   = alloc16((size_t)B * 1600);
  u16t* x_bf   = alloc16((size_t)B * 1200);
  u16t* hid_bf = alloc16((size_t)B * 256);

  auto cvt = [&](const float* s, u16t* d, size_t n) {
    long n4 = (long)(n / 4);
    int blocks = (int)((n4 + 255) / 256);
    if (blocks > 4096) blocks = 4096;
    hipLaunchKernelGGL(cvt_f32_to_bf16, dim3(blocks), dim3(256), 0, stream, s, d, n4);
  };

  // one-time (per call) bf16 conversions; Wm0+Wm1 bf16 = 153.6MB -> fits 192MB L2
  cvt(input_v, v_bf, (size_t)B * 768);
  cvt(input_q, q_bf, (size_t)B * 2400);
  cvt(Wv,  Wv_bf,  (size_t)1200 * 768);
  cvt(Wq,  Wq_bf,  (size_t)1200 * 2400);
  cvt(W0,  W0_bf,  (size_t)1600 * 1200);
  cvt(W1,  W1_bf,  (size_t)1600 * 1200);
  cvt(Wm0, Wm0_bf, (size_t)24000 * 1600);
  cvt(Wm1, Wm1_bf, (size_t)24000 * 1600);
  cvt(Wo,  Wo_bf,  (size_t)1200 * 1600);
  for (int q = 0; q < 4; ++q)
    cvt((const float*)d_in[17 + 4 * q], cW1_bf[q], (size_t)256 * 1200);

  // projections with tanh
  hipLaunchKernelGGL((gemm_bf16_wmma<true>),  dim3(B / BM, (1200 + BN - 1) / BN), dim3(256), 0, stream,
                     v_bf, Wv_bf, bv, xv_bf, B, 1200, 768);
  hipLaunchKernelGGL((gemm_bf16_wmma<true>),  dim3(B / BM, (1200 + BN - 1) / BN), dim3(256), 0, stream,
                     q_bf, Wq_bf, bq, xq_bf, B, 1200, 2400);
  // Mutan input linears
  hipLaunchKernelGGL((gemm_bf16_wmma<false>), dim3(B / BM, 1600 / BN), dim3(256), 0, stream,
                     xv_bf, W0_bf, b0, h0_bf, B, 1600, 1200);
  hipLaunchKernelGGL((gemm_bf16_wmma<false>), dim3(B / BM, 1600 / BN), dim3(256), 0, stream,
                     xq_bf, W1_bf, b1, h1_bf, B, 1600, 1200);
  // fused rank-15 bilinear core (630 GFLOP, the roofline-dominant step)
  hipLaunchKernelGGL(mutan_fused_wmma, dim3(B / BM, 1600 / BN), dim3(256), 0, stream,
                     h0_bf, h1_bf, Wm0_bf, Wm1_bf, bm0, bm1, z_bf);
  // output linear
  hipLaunchKernelGGL((gemm_bf16_wmma<false>), dim3(B / BM, (1200 + BN - 1) / BN), dim3(256), 0, stream,
                     z_bf, Wo_bf, bo, x_bf, B, 1200, 1600);

  // classifiers: zero out, then per-qt hidden GEMM (WMMA) + masked tiny head
  {
    long n = (long)B * 95;
    hipLaunchKernelGGL(zero_f32, dim3((int)((n + 255) / 256)), dim3(256), 0, stream, out, n);
  }
  const int NA[4] = {2, 2, 5, 88};
  const int LO[4] = {0, 0, 2, 7};
  for (int q = 0; q < 4; ++q) {
    hipLaunchKernelGGL((gemm_bf16_wmma<true>), dim3(B / BM, 256 / BN), dim3(256), 0, stream,
                       x_bf, cW1_bf[q], (const float*)d_in[18 + 4 * q], hid_bf, B, 256, 1200);
    hipLaunchKernelGGL(classifier_head, dim3(B), dim3(128), 0, stream,
                       hid_bf, (const float*)d_in[19 + 4 * q], (const float*)d_in[20 + 4 * q],
                       qtype, q, LO[q], NA[q], out);
  }
}
